// attLIN_41979010351762
// MI455X (gfx1250) — matrix operand order repaired and checked by reading, not compiled, not run on hardware
//
#include <hip/hip_runtime.h>
#include <hip/hip_bf16.h>

// ---------------------------------------------------------------------------
// MI455X (gfx1250) fused LinearAttention block.
// Memory-bound: bf16 WMMA GEMMs (f32 accum), GN folded to per-channel
// scale/shift applied during LDS staging (ping-pong buffers), attention
// context folded into the output-projection weights.  LDS tiles stored in
// WMMA-fragment order so B-fragments load as 2x ds_load_b128.
// ---------------------------------------------------------------------------

typedef __attribute__((ext_vector_type(16))) __bf16 v16bf;
typedef __attribute__((ext_vector_type(8)))  __bf16 v8bf;
typedef __attribute__((ext_vector_type(8)))  float  v8f;
typedef __bf16 bf16_t;

#define BQ 8
#define TDIM 8192

__device__ __forceinline__ float dev_exp(float x) { return __expf(x); }

// ---------------------------------------------------------------------------
// GroupNorm stats: one block per (b, group). st[b*64+g*2] = mean, +1 = rstd.
// ---------------------------------------------------------------------------
__global__ void k_gn_stats(const float* __restrict__ x, long bstride,
                           int C, int T, float* __restrict__ st) {
  const int b = blockIdx.x >> 5;
  const int g = blockIdx.x & 31;
  const int gs = C >> 5;
  const float* p = x + (long)b * bstride + (long)g * gs * T;
  const long n = (long)gs * T;
  float s = 0.f, s2 = 0.f;
  for (long i = threadIdx.x; i < n; i += blockDim.x) {
    float v = p[i];
    s += v; s2 += v * v;
  }
  __shared__ float as[256], as2[256];
  as[threadIdx.x] = s; as2[threadIdx.x] = s2;
  __syncthreads();
  for (int off = 128; off > 0; off >>= 1) {
    if ((int)threadIdx.x < off) {
      as[threadIdx.x]  += as[threadIdx.x + off];
      as2[threadIdx.x] += as2[threadIdx.x + off];
    }
    __syncthreads();
  }
  if (threadIdx.x == 0) {
    float mu  = as[0] / (float)n;
    float var = as2[0] / (float)n - mu * mu;
    st[(long)blockIdx.x * 2]     = mu;
    st[(long)blockIdx.x * 2 + 1] = rsqrtf(var + 1e-5f);
  }
}

// ---------------------------------------------------------------------------
// Expand GN stats to per-channel affine: coeff[b][c] = (scale, shift)
// ---------------------------------------------------------------------------
__global__ void k_gn_coeffs(const float* __restrict__ st,
                            const float* __restrict__ gamma,
                            const float* __restrict__ beta,
                            int C, float* __restrict__ coeff) {
  const int idx = blockIdx.x * 256 + threadIdx.x;   // b*C + c
  const int c = idx % C;
  const int b = idx / C;
  const int g = c / (C >> 5);
  const float mu = st[(long)b * 64 + 2 * g];
  const float rs = st[(long)b * 64 + 2 * g + 1];
  const float scl = rs * gamma[c];
  coeff[(long)idx * 2]     = scl;
  coeff[(long)idx * 2 + 1] = beta[c] - mu * scl;
}

// ---------------------------------------------------------------------------
// Generic GEMM: Y[b] = W[b] @ affine(X[b])  (+bias/silu/residual epilogue)
//   Compile-time: HAS_CF (per-channel affine on X), EPI bits 1=bias 2=silu
//   4=residual.  Block 256 thr (8 waves). Tile M=128, N=64, K-step 32.
//   Ping-pong LDS tile in fragment order lb[buf][n][k] (96B row stride).
// ---------------------------------------------------------------------------
template <int EPI, bool HAS_CF>
__global__ void k_gemm_aff_bf16(const float* __restrict__ W, long wbs,
                                const float* __restrict__ X, long xbs,
                                float* __restrict__ Y, long ybs,
                                const float* __restrict__ coeff,
                                const float* __restrict__ bias,
                                const float* __restrict__ res, long rbs,
                                int M, int C, int T) {
  const int b    = blockIdx.y;
  const int nblk = T >> 6;
  const int mblk = blockIdx.x / nblk;
  const int tblk = blockIdx.x % nblk;
  const int m0 = mblk * 128;
  const int t0 = tblk * 64;
  const int tid  = threadIdx.x;
  const int lane = tid & 31;
  const int wid  = tid >> 5;

  const float* Wb = W + (long)b * wbs;
  const float* Xb = X + (long)b * xbs;
  float*       Yb = Y + (long)b * ybs;

  __shared__ bf16_t lb[2][64][48];   // ping-pong, [n][k], 96B row stride

  const v8f z = {0.f,0.f,0.f,0.f,0.f,0.f,0.f,0.f};
  v8f acc[4] = {z, z, z, z};

  const int mrow  = m0 + wid * 16;
  const int mr16  = lane & 15;
  const int khalf = (lane >> 4) * 8;
  const int kb    = (lane >> 4) * 16;
  // staging assignment: this thread packs 8 K-values for one column
  const int scc = tid & 63;          // column within tile
  const int srh = (tid >> 6) * 8;    // k-subrange base (0,8,16,24)

  // running pointers (all loop loads become base + immediate offset)
  const float* xp  = Xb + (long)srh * T + t0 + scc;
  const float* cfp = HAS_CF ? (coeff + (long)b * C * 2 + 2 * srh) : nullptr;
  const float* wp  = Wb + (long)(mrow + mr16) * C;

  // ---- prologue: stage K-step 0 into buffer 0
  {
    v8bf pk;
#pragma unroll
    for (int j = 0; j < 8; ++j) {
      float v = xp[(long)j * T];
      if (HAS_CF) {
        const float2 cs = *(const float2*)(cfp + 2 * j);
        v = fmaf(v, cs.x, cs.y);
      }
      pk[j] = (bf16_t)v;
    }
    *(v8bf*)(&lb[0][scc][srh]) = pk;
  }
  xp += 32L * T;
  if (HAS_CF) cfp += 64;
  __syncthreads();

  int cur = 0;
  for (int k0 = 0; k0 < C; k0 += 32) {
    // ---- stage next K-step into the other buffer (overlaps with compute)
    if (k0 + 32 < C) {
      v8bf pk;
#pragma unroll
      for (int j = 0; j < 8; ++j) {
        float v = xp[(long)j * T];
        if (HAS_CF) {
          const float2 cs = *(const float2*)(cfp + 2 * j);
          v = fmaf(v, cs.x, cs.y);
        }
        pk[j] = (bf16_t)v;
      }
      *(v8bf*)(&lb[cur ^ 1][scc][srh]) = pk;
      xp += 32L * T;
      if (HAS_CF) cfp += 64;
    }

    // ---- A fragment: W rows [mrow..mrow+15], cols [k0..k0+31] (4x b128)
    v16bf afrag;
    {
      __builtin_prefetch(wp + 32, 0, 0);
      const float4 w0 = *(const float4*)(wp + khalf);
      const float4 w1 = *(const float4*)(wp + khalf + 4);
      const float4 w2 = *(const float4*)(wp + 16 + khalf);
      const float4 w3 = *(const float4*)(wp + 16 + khalf + 4);
      afrag[0]  = (bf16_t)w0.x; afrag[1]  = (bf16_t)w0.y;
      afrag[2]  = (bf16_t)w0.z; afrag[3]  = (bf16_t)w0.w;
      afrag[4]  = (bf16_t)w1.x; afrag[5]  = (bf16_t)w1.y;
      afrag[6]  = (bf16_t)w1.z; afrag[7]  = (bf16_t)w1.w;
      afrag[8]  = (bf16_t)w2.x; afrag[9]  = (bf16_t)w2.y;
      afrag[10] = (bf16_t)w2.z; afrag[11] = (bf16_t)w2.w;
      afrag[12] = (bf16_t)w3.x; afrag[13] = (bf16_t)w3.y;
      afrag[14] = (bf16_t)w3.z; afrag[15] = (bf16_t)w3.w;
    }
    wp += 32;

    // ---- preload all 4 B fragments, then issue 4 WMMAs back-to-back
    v16bf bfr[4];
#pragma unroll
    for (int nt = 0; nt < 4; ++nt) {
      const int n = nt * 16 + (lane & 15);
      const v8bf lo = *(const v8bf*)(&lb[cur][n][khalf]);
      const v8bf hi = *(const v8bf*)(&lb[cur][n][16 + khalf]);
      bfr[nt] = __builtin_shufflevector(
          lo, hi, 0,1,2,3,4,5,6,7,8,9,10,11,12,13,14,15);
    }
#pragma unroll
    for (int nt = 0; nt < 4; ++nt)
      acc[nt] = __builtin_amdgcn_wmma_f32_16x16x32_bf16(
          false, afrag, false, bfr[nt], (short)0, acc[nt], false, false);

    __syncthreads();
    cur ^= 1;
  }

  // ---- epilogue: row = mrow + (lane>=16 ? 8 : 0) + r, col = lane&15
  const int mbase = mrow + (lane >> 4) * 8;
  const int nlane = lane & 15;
#pragma unroll
  for (int nt = 0; nt < 4; ++nt) {
    const int t = t0 + nt * 16 + nlane;
#pragma unroll
    for (int r = 0; r < 8; ++r) {
      const int m = mbase + r;
      float v = acc[nt][r];
      if (EPI & 1) v += bias[m];
      if (EPI & 2) v = v * (1.f / (1.f + dev_exp(-v)));   // SiLU
      if (EPI & 4) v += res[(long)b * rbs + (long)m * T + t];
      Yb[(long)m * T + t] = v;
    }
  }
}

// ---------------------------------------------------------------------------
// q softmax over the 32 features of each head (axis=d), times scale; in place.
// ---------------------------------------------------------------------------
__global__ void k_softmax_d(float* __restrict__ q, long bstride, int T,
                            float scale) {
  const long idx = (long)blockIdx.x * blockDim.x + threadIdx.x;
  const int  t = (int)(idx % T);
  const long r = idx / T;
  const int  h = (int)(r & 7);
  const int  b = (int)(r >> 3);
  float* p = q + (long)b * bstride + (long)h * 32 * T + t;
  float vals[32];
  float m = -1e30f;
#pragma unroll
  for (int d = 0; d < 32; ++d) { vals[d] = p[(long)d * T]; m = fmaxf(m, vals[d]); }
  float s = 0.f;
#pragma unroll
  for (int d = 0; d < 32; ++d) { vals[d] = dev_exp(vals[d] - m); s += vals[d]; }
  const float inv = scale / s;
#pragma unroll
  for (int d = 0; d < 32; ++d) p[(long)d * T] = vals[d] * inv;
}

// ---------------------------------------------------------------------------
// Online softmax stats over time for k rows: one block per row (b*256+r).
// ---------------------------------------------------------------------------
__global__ void k_rowstats(const float* __restrict__ src, long bstride,
                           float* __restrict__ omax, float* __restrict__ oden,
                           int rowsPerB, int T) {
  const int row = blockIdx.x;
  const int b = row / rowsPerB;
  const int r = row % rowsPerB;
  const float* p = src + (long)b * bstride + (long)r * T;
  float m = -1e30f, s = 0.f;
  for (int t = threadIdx.x; t < T; t += blockDim.x) {
    const float v = p[t];
    if (v > m) { s = s * dev_exp(m - v) + 1.f; m = v; }
    else       { s += dev_exp(v - m); }
  }
  __shared__ float sm[256], ss[256];
  sm[threadIdx.x] = m; ss[threadIdx.x] = s;
  __syncthreads();
  for (int off = 128; off > 0; off >>= 1) {
    if ((int)threadIdx.x < off) {
      const float m1 = sm[threadIdx.x], s1 = ss[threadIdx.x];
      const float m2 = sm[threadIdx.x + off], s2 = ss[threadIdx.x + off];
      const float mm = fmaxf(m1, m2);
      sm[threadIdx.x] = mm;
      ss[threadIdx.x] = s1 * dev_exp(m1 - mm) + s2 * dev_exp(m2 - mm);
    }
    __syncthreads();
  }
  if (threadIdx.x == 0) { omax[row] = sm[0]; oden[row] = ss[0]; }
}

// ---------------------------------------------------------------------------
// ctx[b,h,d,e] = (sum_t exp(k[d,t]-max[d]) * v[e,t]) / den[d]
// One block (8 waves) per (b,h).  WMMA over K=T, cross-wave LDS ds_add_f32.
// ---------------------------------------------------------------------------
__global__ void k_ctx_gemm(const float* __restrict__ K, long kbs,
                           const float* __restrict__ V, long vbs,
                           const float* __restrict__ kmax,
                           const float* __restrict__ kden,
                           float* __restrict__ ctx, int T) {
  const int b = blockIdx.x >> 3;
  const int h = blockIdx.x & 7;
  const float* Kb = K + (long)b * kbs + (long)h * 32 * T;
  const float* Vb = V + (long)b * vbs + (long)h * 32 * T;
  const float* mx = kmax + (b * 8 + h) * 32;
  const float* dn = kden + (b * 8 + h) * 32;
  const int lane = threadIdx.x & 31;
  const int wid  = threadIdx.x >> 5;

  __shared__ float cacc[32][32];
  for (int i = threadIdx.x; i < 1024; i += 256) ((float*)cacc)[i] = 0.f;
  __syncthreads();

  const v8f z = {0.f,0.f,0.f,0.f,0.f,0.f,0.f,0.f};
  v8f acc[2][2] = {{z, z}, {z, z}};
  const int m  = lane & 15;
  const int kh = (lane >> 4);

  for (int t0 = wid * 32; t0 < T; t0 += 8 * 32) {
    v16bf a[2], bb[2];
#pragma unroll
    for (int dt = 0; dt < 2; ++dt) {
      const int row = dt * 16 + m;
      const float* kr = Kb + (long)row * T + t0;
      const float mxr = mx[row];
      const float4 k0v = *(const float4*)(kr + kh * 8);
      const float4 k1v = *(const float4*)(kr + kh * 8 + 4);
      const float4 k2v = *(const float4*)(kr + 16 + kh * 8);
      const float4 k3v = *(const float4*)(kr + 16 + kh * 8 + 4);
      a[dt][0]  = (bf16_t)dev_exp(k0v.x - mxr); a[dt][1]  = (bf16_t)dev_exp(k0v.y - mxr);
      a[dt][2]  = (bf16_t)dev_exp(k0v.z - mxr); a[dt][3]  = (bf16_t)dev_exp(k0v.w - mxr);
      a[dt][4]  = (bf16_t)dev_exp(k1v.x - mxr); a[dt][5]  = (bf16_t)dev_exp(k1v.y - mxr);
      a[dt][6]  = (bf16_t)dev_exp(k1v.z - mxr); a[dt][7]  = (bf16_t)dev_exp(k1v.w - mxr);
      a[dt][8]  = (bf16_t)dev_exp(k2v.x - mxr); a[dt][9]  = (bf16_t)dev_exp(k2v.y - mxr);
      a[dt][10] = (bf16_t)dev_exp(k2v.z - mxr); a[dt][11] = (bf16_t)dev_exp(k2v.w - mxr);
      a[dt][12] = (bf16_t)dev_exp(k3v.x - mxr); a[dt][13] = (bf16_t)dev_exp(k3v.y - mxr);
      a[dt][14] = (bf16_t)dev_exp(k3v.z - mxr); a[dt][15] = (bf16_t)dev_exp(k3v.w - mxr);
    }
#pragma unroll
    for (int et = 0; et < 2; ++et) {
      const int col = et * 16 + (lane & 15);
      const float* vr = Vb + (long)col * T + t0 + kh * 8;
      const float4 v0 = *(const float4*)(vr);
      const float4 v1 = *(const float4*)(vr + 4);
      const float4 v2 = *(const float4*)(vr + 16);
      const float4 v3 = *(const float4*)(vr + 20);
      bb[et][0]  = (bf16_t)v0.x; bb[et][1]  = (bf16_t)v0.y;
      bb[et][2]  = (bf16_t)v0.z; bb[et][3]  = (bf16_t)v0.w;
      bb[et][4]  = (bf16_t)v1.x; bb[et][5]  = (bf16_t)v1.y;
      bb[et][6]  = (bf16_t)v1.z; bb[et][7]  = (bf16_t)v1.w;
      bb[et][8]  = (bf16_t)v2.x; bb[et][9]  = (bf16_t)v2.y;
      bb[et][10] = (bf16_t)v2.z; bb[et][11] = (bf16_t)v2.w;
      bb[et][12] = (bf16_t)v3.x; bb[et][13] = (bf16_t)v3.y;
      bb[et][14] = (bf16_t)v3.z; bb[et][15] = (bf16_t)v3.w;
    }
#pragma unroll
    for (int dt = 0; dt < 2; ++dt)
#pragma unroll
      for (int et = 0; et < 2; ++et)
        acc[dt][et] = __builtin_amdgcn_wmma_f32_16x16x32_bf16(
            false, a[dt], false, bb[et], (short)0, acc[dt][et], false, false);
  }

#pragma unroll
  for (int dt = 0; dt < 2; ++dt)
#pragma unroll
    for (int et = 0; et < 2; ++et) {
      const int e = et * 16 + (lane & 15);
#pragma unroll
      for (int r = 0; r < 8; ++r) {
        const int d = dt * 16 + (lane >> 4) * 8 + r;
        atomicAdd(&cacc[d][e], acc[dt][et][r]);    // LDS ds_add_f32
      }
    }
  __syncthreads();
  float* outp = ctx + (long)(b * 8 + h) * 1024;
  for (int i = threadIdx.x; i < 1024; i += 256) {
    const int d = i >> 5;
    outp[i] = cacc[d][i & 31] / dn[d];
  }
}

// ---------------------------------------------------------------------------
// Fold ctx into projection weight:
//   weff[b][o][h*32+d] = sum_e Wproj[o][h*32+e] * ctx[b,h,d,e]
// ---------------------------------------------------------------------------
__global__ void k_weff(const float* __restrict__ Wproj,
                       const float* __restrict__ ctx,
                       float* __restrict__ weff) {
  const int idx = blockIdx.x * 256 + threadIdx.x;   // b*65536 + o*256 + col
  const int col = idx & 255;
  const int o   = (idx >> 8) & 255;
  const int b   = idx >> 16;
  const int h = col >> 5, d = col & 31;
  const float* wr = Wproj + (long)o * 256 + h * 32;
  const float* cr = ctx + (long)((b * 8 + h) * 32 + d) * 32;
  float s = 0.f;
#pragma unroll
  for (int e = 0; e < 32; ++e) s += wr[e] * cr[e];
  weff[idx] = s;
}

// ---------------------------------------------------------------------------
// out = GroupNorm(y; stats, gamma, beta) + res     (elementwise)
// ---------------------------------------------------------------------------
__global__ void k_gn_apply_res(const float* __restrict__ y,
                               const float* __restrict__ st,
                               const float* __restrict__ gamma,
                               const float* __restrict__ beta,
                               const float* __restrict__ res,
                               float* __restrict__ out, int C, int T) {
  const long idx = (long)blockIdx.x * blockDim.x + threadIdx.x;
  const long ct = (long)C * T;
  const int  b = (int)(idx / ct);
  const long rem = idx % ct;
  const int  c = (int)(rem / T);
  const int  g = c / (C >> 5);
  const float mu = st[(long)b * 64 + 2 * g];
  const float rs = st[(long)b * 64 + 2 * g + 1];
  out[idx] = (y[idx] - mu) * rs * gamma[c] + beta[c] + res[idx];
}

// ---------------------------------------------------------------------------
// Host launcher
// ---------------------------------------------------------------------------
extern "C" void kernel_launch(void* const* d_in, const int* in_sizes, int n_in,
                              void* d_out, int out_size, void* d_ws, size_t ws_size,
                              hipStream_t stream) {
  (void)in_sizes; (void)n_in; (void)out_size; (void)ws_size;
  const int T = TDIM, Bn = BQ;
  const long sA = 768L * T;          // mega-buffer per-batch stride

  const float* x     = (const float*)d_in[0];
  const float* c     = (const float*)d_in[1];
  const float* W_qkv = (const float*)d_in[2];
  const float* W_so  = (const float*)d_in[3];
  const float* b_so  = (const float*)d_in[4];
  const float* sa_g  = (const float*)d_in[5];
  const float* sa_b  = (const float*)d_in[6];
  const float* W_cq  = (const float*)d_in[7];
  const float* W_ckv = (const float*)d_in[8];
  const float* W_co  = (const float*)d_in[9];
  const float* b_co  = (const float*)d_in[10];
  const float* ca_g  = (const float*)d_in[11];
  const float* ca_b  = (const float*)d_in[12];
  const float* W_m1  = (const float*)d_in[13];
  const float* b_m1  = (const float*)d_in[14];
  const float* W_m2  = (const float*)d_in[15];
  const float* b_m2  = (const float*)d_in[16];
  const float* nm_g  = (const float*)d_in[17];
  const float* nm_b  = (const float*)d_in[18];
  const float* nm1_g = (const float*)d_in[19];
  const float* nm1_b = (const float*)d_in[20];
  const float* nm2_g = (const float*)d_in[21];
  const float* nm2_b = (const float*)d_in[22];
  const float* nm3_g = (const float*)d_in[23];
  const float* nm3_b = (const float*)d_in[24];
  float* out = (float*)d_out;

  float* ws   = (float*)d_ws;
  float* bufA = ws;                             // [B,768,T] multi-purpose
  float* bufY = bufA + (size_t)Bn * sA;         // [B,256,T]
  float* bufX1 = bufY + (size_t)Bn * 256 * T;   // [B,256,T]
  float* stp  = bufX1 + (size_t)Bn * 256 * T;   // 6 stat sets x 512
  float* st0 = stp, *st_sa = stp + 512, *st1 = stp + 1024, *st3 = stp + 1536,
       * st_ca = stp + 2048, *st2 = stp + 2560;
  float* kmax = stp + 3072;                     // [B*256]
  float* kden = kmax + 2048;                    // [B*256]
  float* ctx  = kden + 2048;                    // [B,8,32,32]
  float* weff = ctx + 65536;                    // [B,256,256]
  float* cf0  = weff + 524288;                  // [B,256,2] GN(x)
  float* cf1  = cf0 + 4096;                     // [B,256,2] GN(x1)
  float* cf2  = cf1 + 4096;                     // [B,256,2] GN(x2)
  float* cf3  = cf2 + 4096;                     // [B,512,2] GN(c)

  const float scale = 0.17677669529663689f;     // 32^-0.5
  const dim3 blk(256);
  const int nblkT = T / 64;

  // ---------------- self linear attention ----------------
  k_gn_stats<<<256, blk, 0, stream>>>(x, 256L * T, 256, T, st0);
  k_gn_coeffs<<<(Bn * 256) / 256, blk, 0, stream>>>(st0, nm_g, nm_b, 256, cf0);
  k_gemm_aff_bf16<0, true><<<dim3(6 * nblkT, Bn), blk, 0, stream>>>(
      W_qkv, 0, x, 256L * T, bufA, sA, cf0, nullptr, nullptr, 0, 768, 256, T);
  k_softmax_d<<<(Bn * 8 * T) / 256, blk, 0, stream>>>(bufA, sA, T, scale);
  k_rowstats<<<Bn * 256, blk, 0, stream>>>(bufA + 256L * T, sA, kmax, kden, 256, T);
  k_ctx_gemm<<<Bn * 8, blk, 0, stream>>>(bufA + 256L * T, sA,
                                         bufA + 512L * T, sA, kmax, kden, ctx, T);
  k_weff<<<(Bn * 256 * 256) / 256, blk, 0, stream>>>(W_so, ctx, weff);
  k_gemm_aff_bf16<1, false><<<dim3(2 * nblkT, Bn), blk, 0, stream>>>(
      weff, 65536, bufA, sA, bufY, 256L * T, nullptr, b_so, nullptr, 0,
      256, 256, T);
  k_gn_stats<<<256, blk, 0, stream>>>(bufY, 256L * T, 256, T, st_sa);
  k_gn_apply_res<<<(int)((Bn * 256L * T) / 256), blk, 0, stream>>>(
      bufY, st_sa, sa_g, sa_b, x, bufX1, 256, T);

  // ---------------- cross linear attention ----------------
  k_gn_stats<<<256, blk, 0, stream>>>(bufX1, 256L * T, 256, T, st1);
  k_gn_stats<<<256, blk, 0, stream>>>(c, 512L * T, 512, T, st3);
  k_gn_coeffs<<<(Bn * 256) / 256, blk, 0, stream>>>(st1, nm1_g, nm1_b, 256, cf1);
  k_gn_coeffs<<<(Bn * 512) / 256, blk, 0, stream>>>(st3, nm3_g, nm3_b, 512, cf3);
  k_gemm_aff_bf16<0, true><<<dim3(2 * nblkT, Bn), blk, 0, stream>>>(
      W_cq, 0, bufX1, 256L * T, bufA + 512L * T, sA, cf1, nullptr, nullptr, 0,
      256, 256, T);
  k_gemm_aff_bf16<0, true><<<dim3(4 * nblkT, Bn), blk, 0, stream>>>(
      W_ckv, 0, c, 512L * T, bufA, sA, cf3, nullptr, nullptr, 0,
      512, 512, T);
  k_softmax_d<<<(Bn * 8 * T) / 256, blk, 0, stream>>>(bufA + 512L * T, sA, T, scale);
  k_rowstats<<<Bn * 256, blk, 0, stream>>>(bufA, sA, kmax, kden, 256, T);
  k_ctx_gemm<<<Bn * 8, blk, 0, stream>>>(bufA, sA, bufA + 256L * T, sA,
                                         kmax, kden, ctx, T);
  k_weff<<<(Bn * 256 * 256) / 256, blk, 0, stream>>>(W_co, ctx, weff);
  k_gemm_aff_bf16<1, false><<<dim3(2 * nblkT, Bn), blk, 0, stream>>>(
      weff, 65536, bufA + 512L * T, sA, bufY, 256L * T, nullptr, b_co,
      nullptr, 0, 256, 256, T);
  k_gn_stats<<<256, blk, 0, stream>>>(bufY, 256L * T, 256, T, st_ca);
  k_gn_apply_res<<<(int)((Bn * 256L * T) / 256), blk, 0, stream>>>(
      bufY, st_ca, ca_g, ca_b, bufX1, out, 256, T);     // x2 -> d_out

  // ---------------- SiLU MLP ----------------
  k_gn_stats<<<256, blk, 0, stream>>>(out, 256L * T, 256, T, st2);
  k_gn_coeffs<<<(Bn * 256) / 256, blk, 0, stream>>>(st2, nm2_g, nm2_b, 256, cf2);
  k_gemm_aff_bf16<1 | 2, true><<<dim3(5 * nblkT, Bn), blk, 0, stream>>>(
      W_m1, 0, out, 256L * T, bufA, sA, cf2, b_m1, nullptr, 0,
      640, 256, T);                                     // bias + SiLU
  k_gemm_aff_bf16<1 | 4, false><<<dim3(2 * nblkT, Bn), blk, 0, stream>>>(
      W_m2, 0, bufA, sA, out, 256L * T, nullptr, b_m2, out, 256L * T,
      256, 640, T);                                     // bias + residual
}
